// LoRALinear_89721866814276
// MI455X (gfx1250) — matrix operand order repaired and checked by reading, not compiled, not run on hardware
//
#include <hip/hip_runtime.h>

typedef __attribute__((ext_vector_type(16))) _Float16 v16h;
typedef __attribute__((ext_vector_type(8)))  _Float16 v8h;
typedef __attribute__((ext_vector_type(4)))  _Float16 v4h;
typedef __attribute__((ext_vector_type(8)))  float    v8f;

#define IN_F   4096
#define OUT_F  4096
#define RANK   16
#define SCALING 2.0f
#define M_TOT  8192   // B*S = 4*2048

// ---------------------------------------------------------------------------
// Kernel 1: convert activations f32 -> f16  (8192x4096)
// ---------------------------------------------------------------------------
__global__ __launch_bounds__(256) void cvt_x_f16(const float* __restrict__ x,
                                                 _Float16* __restrict__ xh) {
    size_t i = ((size_t)blockIdx.x * blockDim.x + threadIdx.x) * 4;
    float4 v = *(const float4*)(x + i);
    v4h h;
    h.x = (_Float16)v.x; h.y = (_Float16)v.y;
    h.z = (_Float16)v.z; h.w = (_Float16)v.w;
    *(v4h*)(xh + i) = h;
}

// ---------------------------------------------------------------------------
// Kernel 2: fold base scale + LoRA into one f16 weight matrix
//   Weff[o,i] = scale[o]*signs[o,i] + 2.0 * sum_r lora_B[o,r]*lora_A[r,i]
// ---------------------------------------------------------------------------
__global__ __launch_bounds__(256) void prep_weights(const float* __restrict__ signs,
                                                    const float* __restrict__ scale,
                                                    const float* __restrict__ lA,
                                                    const float* __restrict__ lB,
                                                    _Float16* __restrict__ wh) {
    const int o = blockIdx.x;
    const float sc = scale[o];
    float bvec[RANK];
#pragma unroll
    for (int r = 0; r < RANK; ++r)
        bvec[r] = lB[(size_t)o * RANK + r] * SCALING;

    for (int i = threadIdx.x; i < IN_F; i += 256) {
        float acc = sc * signs[(size_t)o * IN_F + i];
#pragma unroll
        for (int r = 0; r < RANK; ++r)
            acc += bvec[r] * lA[(size_t)r * IN_F + i];
        wh[(size_t)o * IN_F + i] = (_Float16)acc;
    }
}

// ---------------------------------------------------------------------------
// Kernel 3: f16 WMMA GEMM with f32 accumulate, async-to-LDS double buffering
//   out[M,N] = Xh[M,K] * Weff[N,K]^T     (both operands K-contiguous)
// Block tile 128x128, 8 waves, each wave 32x64 (2x4 tiles of 16x16).
// ---------------------------------------------------------------------------
#define BM 128
#define BN 128
#define BK 32
#define LDSS 40            // row stride in halves (32 + 8 pad -> 80B)
#define TILE_H (BM * LDSS) // halves per (matrix, buffer) = 5120

// Issue 4 async memory->LDS 16B transfers (one tile's worth for this thread).
__device__ __forceinline__ void issue4(unsigned l0, unsigned long long g0,
                                       unsigned l1, unsigned long long g1,
                                       unsigned l2, unsigned long long g2,
                                       unsigned l3, unsigned long long g3) {
    asm volatile("global_load_async_to_lds_b128 %0, %1, off" :: "v"(l0), "v"(g0) : "memory");
    asm volatile("global_load_async_to_lds_b128 %0, %1, off" :: "v"(l1), "v"(g1) : "memory");
    asm volatile("global_load_async_to_lds_b128 %0, %1, off" :: "v"(l2), "v"(g2) : "memory");
    asm volatile("global_load_async_to_lds_b128 %0, %1, off" :: "v"(l3), "v"(g3) : "memory");
}

__device__ __forceinline__ void compute_tile(const _Float16* __restrict__ pA,
                                             const _Float16* __restrict__ pB,
                                             int wm, int wn, int lrow,
                                             int kbA, int kbB,
                                             v8f acc[2][4]) {
    v16h afrag[2], bfrag[4];
#pragma unroll
    for (int t = 0; t < 2; ++t) {
        const _Float16* p = pA + (wm + t * 16 + lrow) * LDSS + kbA;
        v8h lo = *(const v8h*)p;
        v8h hi = *(const v8h*)(p + 16);
        afrag[t] = __builtin_shufflevector(lo, hi, 0,1,2,3,4,5,6,7,
                                                   8,9,10,11,12,13,14,15);
    }
#pragma unroll
    for (int u = 0; u < 4; ++u) {
        const _Float16* p = pB + (wn + u * 16 + lrow) * LDSS + kbB;
        v8h lo = *(const v8h*)p;
        v8h hi = *(const v8h*)(p + 16);
        bfrag[u] = __builtin_shufflevector(lo, hi, 0,1,2,3,4,5,6,7,
                                                   8,9,10,11,12,13,14,15);
    }
#pragma unroll
    for (int t = 0; t < 2; ++t)
#pragma unroll
        for (int u = 0; u < 4; ++u)
            acc[t][u] = __builtin_amdgcn_wmma_f32_16x16x32_f16(
                false, afrag[t], false, bfrag[u],
                (short)0, acc[t][u], false, false);
}

__global__ __launch_bounds__(256) void lora_gemm_f16(const _Float16* __restrict__ A,
                                                     const _Float16* __restrict__ W,
                                                     float* __restrict__ out) {
    __shared__ _Float16 sA[2][TILE_H];
    __shared__ _Float16 sB[2][TILE_H];

    const int tid  = threadIdx.x;
    const int wave = tid >> 5;
    const int lane = tid & 31;
    const int m0 = blockIdx.y * BM;
    const int n0 = blockIdx.x * BN;

    const int wm = (wave & 3) * 32;   // 4 waves along M
    const int wn = (wave >> 2) * 64;  // 2 waves along N

    const int lrow   = lane & 15;
    const int hihalf = lane >> 4;
    const int kbA = hihalf * 8;       // A frag: K {kb..kb+7, kb+16..kb+23}
    const int kbB = hihalf * 8;       // B frag: K {kb..kb+7, kb+16..kb+23}

    // --- staging assignment: thread owns 16B chunks c=tid and c=tid+256 of
    // each 128x32 tile (chunk c: row = c>>2, 16B-sub = c&3; chunk tid+256 is
    // +64 rows). Global pointers advance by BK halves (64B) per K-tile.
    const int crow = tid >> 2;        // 0..63
    const int csub = tid & 3;         // 0..3
    unsigned long long gA0 = (unsigned long long)(A + (size_t)(m0 + crow) * IN_F + csub * 8);
    unsigned long long gA1 = gA0 + (unsigned long long)64 * IN_F * sizeof(_Float16);
    unsigned long long gB0 = (unsigned long long)(W + (size_t)(n0 + crow) * IN_F + csub * 8);
    unsigned long long gB1 = gB0 + (unsigned long long)64 * IN_F * sizeof(_Float16);

    const int lidx0 = crow * LDSS + csub * 8;
    const int lidx1 = lidx0 + 64 * LDSS;
    const unsigned laA[2][2] = {
        { (unsigned)(size_t)&sA[0][lidx0], (unsigned)(size_t)&sA[0][lidx1] },
        { (unsigned)(size_t)&sA[1][lidx0], (unsigned)(size_t)&sA[1][lidx1] } };
    const unsigned laB[2][2] = {
        { (unsigned)(size_t)&sB[0][lidx0], (unsigned)(size_t)&sB[0][lidx1] },
        { (unsigned)(size_t)&sB[1][lidx0], (unsigned)(size_t)&sB[1][lidx1] } };

    v8f zero = {};
    v8f acc[2][4];
#pragma unroll
    for (int t = 0; t < 2; ++t)
#pragma unroll
        for (int u = 0; u < 4; ++u) acc[t][u] = zero;

    const int NT = IN_F / BK;         // 128 K-tiles (even)

    // Prologue: tile 0 -> buffer 0
    issue4(laA[0][0], gA0, laA[0][1], gA1, laB[0][0], gB0, laB[0][1], gB1);
    gA0 += 64; gA1 += 64; gB0 += 64; gB1 += 64;

    for (int kt = 0; kt < NT; kt += 2) {
        // ---- even step: prefetch tile kt+1 -> buf1, compute tile kt (buf0)
        issue4(laA[1][0], gA0, laA[1][1], gA1, laB[1][0], gB0, laB[1][1], gB1);
        gA0 += 64; gA1 += 64; gB0 += 64; gB1 += 64;
        asm volatile("s_wait_asynccnt 4" ::: "memory");  // tile kt landed
        __syncthreads();
        compute_tile(sA[0], sB[0], wm, wn, lrow, kbA, kbB, acc);
        __syncthreads();

        // ---- odd step: prefetch tile kt+2 -> buf0, compute tile kt+1 (buf1)
        if (kt + 2 < NT) {
            issue4(laA[0][0], gA0, laA[0][1], gA1, laB[0][0], gB0, laB[0][1], gB1);
            gA0 += 64; gA1 += 64; gB0 += 64; gB1 += 64;
            asm volatile("s_wait_asynccnt 4" ::: "memory");
        } else {
            asm volatile("s_wait_asynccnt 0" ::: "memory");
        }
        __syncthreads();
        compute_tile(sA[1], sB[1], wm, wn, lrow, kbA, kbB, acc);
        __syncthreads();
    }

    // Store: C/D layout -> VGPR v holds M = v (lanes 0-15) / M = 8+v (16-31),
    // N = lane%16.
#pragma unroll
    for (int t = 0; t < 2; ++t) {
        const int mbase = m0 + wm + t * 16 + hihalf * 8;
#pragma unroll
        for (int u = 0; u < 4; ++u) {
            const int col = n0 + wn + u * 16 + lrow;
#pragma unroll
            for (int v = 0; v < 8; ++v)
                out[(size_t)(mbase + v) * OUT_F + col] = acc[t][u][v];
        }
    }
}

// ---------------------------------------------------------------------------
extern "C" void kernel_launch(void* const* d_in, const int* in_sizes, int n_in,
                              void* d_out, int out_size, void* d_ws, size_t ws_size,
                              hipStream_t stream) {
    const float* x     = (const float*)d_in[0];  // [4,2048,4096]
    const float* signs = (const float*)d_in[1];  // [4096,4096]
    const float* scale = (const float*)d_in[2];  // [4096]
    const float* lA    = (const float*)d_in[3];  // [16,4096]
    const float* lB    = (const float*)d_in[4];  // [4096,16]
    float* out = (float*)d_out;                  // [4,2048,4096]

    // Workspace layout: Weff f16 (32 MB) | x f16 (64 MB)
    _Float16* wh = (_Float16*)d_ws;
    _Float16* xh = (_Float16*)((char*)d_ws + (size_t)OUT_F * IN_F * sizeof(_Float16));

    // 1) x -> f16
    {
        const size_t n = (size_t)M_TOT * IN_F;           // 33.5M
        dim3 grid((unsigned)(n / (256 * 4)));
        cvt_x_f16<<<grid, 256, 0, stream>>>(x, xh);
    }
    // 2) fold scale + LoRA into f16 weights
    prep_weights<<<dim3(OUT_F), 256, 0, stream>>>(signs, scale, lA, lB, wh);

    // 3) WMMA GEMM
    {
        dim3 grid(OUT_F / BN, M_TOT / BM);               // 32 x 64 blocks
        lora_gemm_f16<<<grid, 256, 0, stream>>>(xh, wh, out);
    }
}
